// SpikingCoreFlow_62629213110827
// MI455X (gfx1250) — matrix operand order repaired and checked by reading, not compiled, not run on hardware
//
#include <hip/hip_runtime.h>

#define N_CORES 16
#define KDIM    256
#define ADIM    256
#define N_IN    3072
#define BATCH   256
#define N_OUT_  256
#define CYCLES  32
#define POOL    (N_IN + N_CORES*KDIM + 2)   // 7170

typedef __attribute__((ext_vector_type(16))) _Float16 v16h;
typedef __attribute__((ext_vector_type(8)))  _Float16 v8h;
typedef __attribute__((ext_vector_type(8)))  float    v8f;

struct U2 { unsigned x, y; };

static __host__ __device__ inline unsigned rotl32(unsigned x, unsigned n) {
    return (x << n) | (x >> (32u - n));
}

// Threefry-2x32-20 (counter-based RNG, matches JAX's PRNG family)
static __host__ __device__ inline U2 tf2x32(U2 k, U2 c) {
    unsigned ks[3] = { k.x, k.y, k.x ^ k.y ^ 0x1BD11BDAu };
    unsigned x0 = c.x + ks[0], x1 = c.y + ks[1];
    const unsigned r[8] = {13,15,26,6,17,29,16,24};
#pragma unroll
    for (int i = 0; i < 5; ++i) {
#pragma unroll
        for (int j = 0; j < 4; ++j) {
            x0 += x1; x1 = rotl32(x1, r[(i & 1) * 4 + j]); x1 ^= x0;
        }
        x0 += ks[(i + 1) % 3];
        x1 += ks[(i + 2) % 3] + (unsigned)(i + 1);
    }
    U2 o; o.x = x0; o.y = x1; return o;
}

static __host__ __device__ inline float unitf(unsigned b) {
    return (float)(b >> 8) * (1.0f / 16777216.0f);
}

// ---------------------------------------------------------------------------
// One-time prep: convert weights f32 -> f16, zero membrane/buffers/output.
// ---------------------------------------------------------------------------
__global__ __launch_bounds__(256)
void prep_kernel(const float* __restrict__ core_params,
                 _Float16* __restrict__ W16,
                 float* __restrict__ memb,
                 _Float16* __restrict__ bufs,
                 float* __restrict__ out) {
    int i = blockIdx.x * blockDim.x + threadIdx.x;
    if (i < N_CORES * KDIM * ADIM) {
        W16[i]  = (_Float16)core_params[i];
        memb[i] = 0.0f;
        bufs[i] = (_Float16)0.0f;
    }
    if (i < BATCH * N_OUT_) out[i] = 0.0f;
}

// ---------------------------------------------------------------------------
// Per-cycle: gather sig[c][b][a] from {RNG spikes, buffers, zo} in f16.
// Only the 1M gathered entries are evaluated (never the full spike tensor).
// ---------------------------------------------------------------------------
__global__ __launch_bounds__(256)
void gather_sig_kernel(const float* __restrict__ x,
                       const int* __restrict__ src_idx,
                       const _Float16* __restrict__ bufs,
                       _Float16* __restrict__ sig,
                       unsigned k0, unsigned k1) {
    int i = blockIdx.x * blockDim.x + threadIdx.x;   // flat over c*B*A
    int a = i & (ADIM - 1);
    int b = (i >> 8) & (BATCH - 1);
    int c = i >> 16;
    int src = src_idx[c * ADIM + a];
    if (src < N_IN) {
        unsigned flat = (unsigned)((c * BATCH + b) * N_IN + src);
        U2 key; key.x = k0; key.y = k1;
        U2 ctr; ctr.x = 0u; ctr.y = flat;
        float u = unitf(tf2x32(key, ctr).x);
        sig[i] = (_Float16)((u < x[b * N_IN + src]) ? 1.0f : 0.0f);
    } else if (src < N_IN + N_CORES * KDIM) {
        int j = src - N_IN;                 // j = c'*K + k'
        sig[i] = bufs[((j >> 8) * BATCH + b) * KDIM + (j & (KDIM - 1))];
    } else {
        sig[i] = (_Float16)((src == POOL - 1) ? 1.0f : 0.0f);
    }
}

// ---------------------------------------------------------------------------
// Per-cycle per-core GEMM + integrate-and-fire.
// grid (2,16), block 256 (8 waves). Block computes a 256x128 (b x k) tile of
// delta = sig x W^T: each wave owns 32 batch rows (two A fragments) and the
// full 128-column strip, so every B fragment staged through LDS feeds TWO
// v_wmma_f32_16x16x32_f16 ops (16 WMMA : 16 ds_load_b128 per k-step).
// Membrane update + binary buffer write fused in the epilogue.
// ---------------------------------------------------------------------------
__global__ __launch_bounds__(256, 1)
void core_gemm_kernel(const _Float16* __restrict__ sig,
                      const _Float16* __restrict__ W16,
                      float* __restrict__ memb,
                      _Float16* __restrict__ bufs,
                      const float* __restrict__ thresholds,
                      const int* __restrict__ latencies,
                      int cycle) {
    int c = blockIdx.y;
    if (cycle < latencies[c]) return;   // uniform: inactive core leaves state untouched

    int tid  = threadIdx.x;
    int wave = tid >> 5;
    int lane = tid & 31;
    int rbase = wave * 32;              // 32 batch rows per wave (covers all 256)
    int cbase = blockIdx.x * 128;       // 128 k-columns per block

    // W tile staged per k-step: 128 kcols x 32 a-values, padded rows:
    // stride 40 halves = 80 B -> 16 B aligned chunks, bank-spread.
    __shared__ _Float16 ldsB[128][40];

    const int hi   = lane >> 4;
    const int lrow = lane & 15;

    v8f acc0[8] = {};   // rows rbase .. rbase+15
    v8f acc1[8] = {};   // rows rbase+16 .. rbase+31

    const _Float16* sigc = sig + (size_t)c * BATCH * ADIM;
    const _Float16* wc   = W16 + (size_t)c * KDIM * ADIM;

    for (int ks = 0; ks < ADIM / 32; ++ks) {
        int abase = ks * 32;
        __syncthreads();
        {   // cooperative stage: thread t -> kcol row t>>1, half-range (t&1)*16
            int kk   = tid >> 1;
            int part = tid & 1;
            const v8h* g = (const v8h*)(wc + (size_t)(cbase + kk) * ADIM + abase + part * 16);
            v8h g0 = g[0], g1 = g[1];
            *(v8h*)(&ldsB[kk][part * 16])     = g0;
            *(v8h*)(&ldsB[kk][part * 16 + 8]) = g1;
        }
        __syncthreads();

        // Two A fragments (ISA A-layout: lane<16 holds K 0-7,16-23; lane>=16 holds 8-15,24-31)
        v16h afrag0, afrag1;
        {
            const _Float16* ap0 = sigc + (size_t)(rbase + lrow) * ADIM + abase;
            const _Float16* ap1 = sigc + (size_t)(rbase + 16 + lrow) * ADIM + abase;
            v8h a0 = *(const v8h*)(ap0 + hi * 8);
            v8h a1 = *(const v8h*)(ap0 + 16 + hi * 8);
            v8h a2 = *(const v8h*)(ap1 + hi * 8);
            v8h a3 = *(const v8h*)(ap1 + 16 + hi * 8);
#pragma unroll
            for (int q = 0; q < 8; ++q) {
                afrag0[q] = a0[q]; afrag0[8 + q] = a1[q];
                afrag1[q] = a2[q]; afrag1[8 + q] = a3[q];
            }
        }

#pragma unroll
        for (int j = 0; j < 8; ++j) {
            // B fragment (lane<16: K 0-7,16-23; lane>=16: K 8-15,24-31)
            v16h bfrag;
            const _Float16* bp = &ldsB[j * 16 + lrow][hi * 8];
            v8h b0 = *(const v8h*)(bp);
            v8h b1 = *(const v8h*)(bp + 16);
#pragma unroll
            for (int q = 0; q < 8; ++q) { bfrag[q] = b0[q]; bfrag[8 + q] = b1[q]; }

            acc0[j] = __builtin_amdgcn_wmma_f32_16x16x32_f16(
                false, afrag0, false, bfrag, (short)0, acc0[j], false, false);
            acc1[j] = __builtin_amdgcn_wmma_f32_16x16x32_f16(
                false, afrag1, false, bfrag, (short)0, acc1[j], false, false);
        }
    }

    // Fused integrate-and-fire epilogue.
    // C/D layout: vgpr i, lane l -> row = i + 8*(l>>4), col = l&15.
    float thr = thresholds[c];
#pragma unroll
    for (int j = 0; j < 8; ++j) {
#pragma unroll
        for (int i = 0; i < 8; ++i) {
            int col = cbase + j * 16 + lrow;
            {
                int row = rbase + i + hi * 8;
                size_t idx = ((size_t)c * BATCH + row) * KDIM + col;
                float m  = memb[idx] + acc0[j][i];
                bool  sp = m > thr;
                memb[idx] = sp ? 0.0f : m;
                bufs[idx] = (_Float16)(sp ? 1.0f : 0.0f);
            }
            {
                int row = rbase + 16 + i + hi * 8;
                size_t idx = ((size_t)c * BATCH + row) * KDIM + col;
                float m  = memb[idx] + acc1[j][i];
                bool  sp = m > thr;
                memb[idx] = sp ? 0.0f : m;
                bufs[idx] = (_Float16)(sp ? 1.0f : 0.0f);
            }
        }
    }
}

// ---------------------------------------------------------------------------
// Per-cycle: gather output pool through out_idx, accumulate into d_out.
// ---------------------------------------------------------------------------
__global__ __launch_bounds__(256)
void out_accum_kernel(const float* __restrict__ x,
                      const int* __restrict__ out_idx,
                      const _Float16* __restrict__ bufs,
                      float* __restrict__ out,
                      unsigned k0, unsigned k1) {
    int o = threadIdx.x;
    int b = blockIdx.x;
    int src = out_idx[o];
    float val;
    if (src < N_IN) {
        unsigned flat = (unsigned)(b * N_IN + src);
        U2 key; key.x = k0; key.y = k1;
        U2 ctr; ctr.x = 0u; ctr.y = flat;
        val = (unitf(tf2x32(key, ctr).x) < x[b * N_IN + src]) ? 1.0f : 0.0f;
    } else if (src < N_IN + N_CORES * KDIM) {
        int j = src - N_IN;
        val = (float)bufs[((j >> 8) * BATCH + b) * KDIM + (j & (KDIM - 1))];
    } else {
        val = (src == POOL - 1) ? 1.0f : 0.0f;
    }
    out[b * N_OUT_ + o] += val;
}

// ---------------------------------------------------------------------------
extern "C" void kernel_launch(void* const* d_in, const int* in_sizes, int n_in_args,
                              void* d_out, int out_size, void* d_ws, size_t ws_size,
                              hipStream_t stream) {
    const float* x           = (const float*)d_in[0];
    const float* core_params = (const float*)d_in[1];
    const float* thresholds  = (const float*)d_in[2];
    const int*   src_idx     = (const int*)d_in[3];
    const int*   out_idx     = (const int*)d_in[4];
    const int*   latencies   = (const int*)d_in[5];
    float* out = (float*)d_out;

    // Workspace layout (10 MiB total; fully L2-resident on MI455X):
    char* ws = (char*)d_ws;
    _Float16* W16  = (_Float16*)(ws);                       // 2 MiB f16 weights
    float*    memb = (float*)   (ws + (2u << 20));          // 4 MiB membrane
    _Float16* bufs = (_Float16*)(ws + (6u << 20));          // 2 MiB spike buffers
    _Float16* sig  = (_Float16*)(ws + (8u << 20));          // 2 MiB gathered A-operand

    prep_kernel<<<4096, 256, 0, stream>>>(core_params, W16, memb, bufs, out);

    U2 base; base.x = 0u; base.y = 42u;   // jax.random.key(42)
    for (int cyc = 0; cyc < CYCLES; ++cyc) {
        U2 cc; cc.x = 0u; cc.y = (unsigned)cyc;
        U2 kc = tf2x32(base, cc);                 // fold_in(rng, cycle)
        U2 z0; z0.x = 0u; z0.y = 0u;
        U2 z1; z1.x = 0u; z1.y = 1u;
        U2 kspk = tf2x32(kc, z0);                 // fold_in(kc, 0) -> spikes
        U2 kout = tf2x32(kc, z1);                 // fold_in(kc, 1) -> out spikes

        gather_sig_kernel<<<4096, 256, 0, stream>>>(x, src_idx, bufs, sig,
                                                    kspk.x, kspk.y);

        dim3 g(2, N_CORES);
        core_gemm_kernel<<<g, 256, 0, stream>>>(sig, W16, memb, bufs,
                                                thresholds, latencies, cyc);

        out_accum_kernel<<<BATCH, N_OUT_, 0, stream>>>(x, out_idx, bufs, out,
                                                       kout.x, kout.y);
    }
}
